// QuantLinearGPTQ_19782619365453
// MI455X (gfx1250) — matrix operand order repaired and checked by reading, not compiled, not run on hardware
//
#include <hip/hip_runtime.h>
#include <hip/hip_bf16.h>
#include <stdint.h>

typedef _Float16 v8h  __attribute__((ext_vector_type(8)));
typedef _Float16 v16h __attribute__((ext_vector_type(16)));
typedef float    v8f  __attribute__((ext_vector_type(8)));

#define IN_F    4096
#define OUT_F   11008
#define N_GRP   32
#define GSIZE   128
#define ZWORDS  (OUT_F / 8)   // 1376
#define TM 128
#define TN 128
#define TK 32
#define NCHUNK (IN_F / TK)    // 128
#define LDSROW 40             // halves per row: 80B stride, conflict-free for b128

union V16 { v16h v; v8h h[2]; };
union PK4 { uint32_t u[4]; v8h h; };

static __device__ __forceinline__ _Float16 nib_to_h1024(uint32_t nib) {
    // f16 bit pattern 0x6400|n encodes exactly 1024+n for n in [0,15]
    union { uint16_t u; _Float16 h; } c;
    c.u = (uint16_t)(0x6400u | nib);
    return c.h;
}

__global__ __launch_bounds__(256)
void gptq_wmma_f16_kernel(const float* __restrict__ x,
                          const uint32_t* __restrict__ qweight,
                          const uint32_t* __restrict__ qzeros,
                          const _Float16* __restrict__ scales,
                          const float* __restrict__ bias,
                          float* __restrict__ out)
{
    __shared__ _Float16 Als[TM * LDSROW];   // x tile [m][kpos] f16
    __shared__ _Float16 Bls[TN * LDSROW];   // dequant W tile, transposed [n][kpos] f16

    const int t    = threadIdx.x;
    const int lane = t & 31;
    const int wave = t >> 5;
    const int wm   = wave & 3;    // M offset = 32*wm
    const int wn   = wave >> 2;   // N offset = 64*wn

    const int nTile = blockIdx.x * TN;
    const int mTile = blockIdx.y * TM;

    // A staging: 2 threads per row, 16 f32 each
    const int arow = t >> 1;
    const int acol = (t & 1) * 16;
    const float* xrow = x + (size_t)(mTile + arow) * IN_F + acol;

    // B staging: 4 packed-k rows x 128 cols; thread handles 2 cols of one row
    const int br = t >> 6;          // packed-k sub-row 0..3
    const int bc = (t & 63) * 2;    // even n within tile
    const int zshift0 = (bc & 7) * 4;
    const int zshift1 = ((bc + 1) & 7) * 4;

    // WMMA fragment LDS offsets (ISA 7.12.2)
    const int fm    = lane & 15;
    const int ahalf = (lane < 16) ? 0 : 8;
    const int bhalf = (lane < 16) ? 0 : 8;

    v8f zero8 = {};
    v8f acc[2][4];
    #pragma unroll
    for (int i = 0; i < 2; ++i)
        #pragma unroll
        for (int j = 0; j < 4; ++j) acc[i][j] = zero8;

    // ---- software-pipeline staging registers ----
    float4   fx0, fx1, fx2, fx3;     // 16 f32 of x for current chunk
    uint32_t qa, qb;                 // 2 packed words for current chunk
    uint32_t zw;                     // raw zero word for current group
    _Float16 s0h, s1h;               // raw scales for current group

    // prologue: chunk 0 + group 0
    {
        const float4* src = (const float4*)xrow;
        fx0 = src[0]; fx1 = src[1]; fx2 = src[2]; fx3 = src[3];
        const uint32_t* qp = qweight + (size_t)br * OUT_F + (nTile + bc);
        qa = qp[0]; qb = qp[1];
        zw  = qzeros[(unsigned)(nTile + bc) >> 3];
        s0h = scales[nTile + bc];
        s1h = scales[nTile + bc + 1];
    }

    for (int g = 0; g < N_GRP; ++g) {
        // derive packed group constants (exact: 1024+z and native f16 scale)
        const _Float16 zh0 = nib_to_h1024((zw >> zshift0) & 15u);
        const _Float16 zh1 = nib_to_h1024((zw >> zshift1) & 15u);
        v8h zs0, zs1, ss0, ss1;
        #pragma unroll
        for (int j = 0; j < 8; ++j) { zs0[j] = zh0; zs1[j] = zh1; ss0[j] = s0h; ss1[j] = s1h; }

        #pragma unroll
        for (int cc = 0; cc < 4; ++cc) {
            const int ch = g * 4 + cc;

            __syncthreads();   // previous compute done before overwriting LDS

            // ---- stage A: f32 -> f16, k-permuted per 8-block: {0,4,1,5,2,6,3,7} ----
            {
                const float* f = (const float*)&fx0;  // fx0..fx3 contiguous? use explicit array
                float fa[16] = { fx0.x, fx0.y, fx0.z, fx0.w,
                                 fx1.x, fx1.y, fx1.z, fx1.w,
                                 fx2.x, fx2.y, fx2.z, fx2.w,
                                 fx3.x, fx3.y, fx3.z, fx3.w };
                (void)f;
                v8h h0, h1;
                h0[0] = (_Float16)fa[0];  h0[1] = (_Float16)fa[4];
                h0[2] = (_Float16)fa[1];  h0[3] = (_Float16)fa[5];
                h0[4] = (_Float16)fa[2];  h0[5] = (_Float16)fa[6];
                h0[6] = (_Float16)fa[3];  h0[7] = (_Float16)fa[7];
                h1[0] = (_Float16)fa[8];  h1[1] = (_Float16)fa[12];
                h1[2] = (_Float16)fa[9];  h1[3] = (_Float16)fa[13];
                h1[4] = (_Float16)fa[10]; h1[5] = (_Float16)fa[14];
                h1[6] = (_Float16)fa[11]; h1[7] = (_Float16)fa[15];
                *(v8h*)&Als[arow * LDSROW + acol]     = h0;
                *(v8h*)&Als[arow * LDSROW + acol + 8] = h1;
            }

            // ---- stage B: magic-number int4 -> packed f16, transposed [n][kpos] ----
            {
                PK4 p0, p1;
                p0.u[0] = ( qa         & 0x000F000Fu) | 0x64006400u;
                p0.u[1] = ((qa >> 4)   & 0x000F000Fu) | 0x64006400u;
                p0.u[2] = ((qa >> 8)   & 0x000F000Fu) | 0x64006400u;
                p0.u[3] = ((qa >> 12)  & 0x000F000Fu) | 0x64006400u;
                p1.u[0] = ( qb         & 0x000F000Fu) | 0x64006400u;
                p1.u[1] = ((qb >> 4)   & 0x000F000Fu) | 0x64006400u;
                p1.u[2] = ((qb >> 8)   & 0x000F000Fu) | 0x64006400u;
                p1.u[3] = ((qb >> 12)  & 0x000F000Fu) | 0x64006400u;
                v8h w0 = (p0.h - zs0) * ss0;   // v_pk_add_f16(neg) + v_pk_mul_f16
                v8h w1 = (p1.h - zs1) * ss1;
                *(v8h*)&Bls[(bc)     * LDSROW + br * 8] = w0;
                *(v8h*)&Bls[(bc + 1) * LDSROW + br * 8] = w1;
            }

            // ---- prefetch next chunk's globals (hidden behind WMMA compute) ----
            if (!(g == N_GRP - 1 && cc == 3)) {
                const int chn = ch + 1;
                const float4* src = (const float4*)(xrow + chn * TK);
                fx0 = src[0]; fx1 = src[1]; fx2 = src[2]; fx3 = src[3];
                const uint32_t* qp =
                    qweight + (size_t)(chn * 4 + br) * OUT_F + (nTile + bc);
                qa = qp[0]; qb = qp[1];
                if (cc == 3) {   // next chunk starts a new group
                    const int gn = g + 1;
                    zw  = qzeros[(size_t)gn * ZWORDS + ((unsigned)(nTile + bc) >> 3)];
                    s0h = scales[(size_t)gn * OUT_F + nTile + bc];
                    s1h = scales[(size_t)gn * OUT_F + nTile + bc + 1];
                }
            }

            __syncthreads();   // tiles visible before compute

            // ---- compute: 2x4 WMMA 16x16x32 f16 ----
            V16 af[2];
            #pragma unroll
            for (int i = 0; i < 2; ++i) {
                const int row = (wm * 32 + i * 16 + fm) * LDSROW;
                af[i].h[0] = *(const v8h*)&Als[row + ahalf];
                af[i].h[1] = *(const v8h*)&Als[row + ahalf + 16];
            }
            V16 bf[4];
            #pragma unroll
            for (int j = 0; j < 4; ++j) {
                const int col = (wn * 64 + j * 16 + fm) * LDSROW;
                bf[j].h[0] = *(const v8h*)&Bls[col + bhalf];
                bf[j].h[1] = *(const v8h*)&Bls[col + bhalf + 16];
            }
            #pragma unroll
            for (int i = 0; i < 2; ++i)
                #pragma unroll
                for (int j = 0; j < 4; ++j)
                    acc[i][j] = __builtin_amdgcn_wmma_f32_16x16x32_f16(
                        false, af[i].v, false, bf[j].v,
                        (short)0, acc[i][j], false, false);
        }
    }

    // ---- epilogue: C layout (VGPR e -> M = e + 8*(lane>=16), N = lane&15) ----
    const int colOff = lane & 15;
    const int rowOff = (lane >> 4) * 8;
    #pragma unroll
    for (int j = 0; j < 4; ++j) {
        const int n  = nTile + wn * 64 + j * 16 + colOff;
        const float bv = bias[n];
        #pragma unroll
        for (int i = 0; i < 2; ++i) {
            const int m0 = mTile + wm * 32 + i * 16 + rowOff;
            #pragma unroll
            for (int e = 0; e < 8; ++e)
                out[(size_t)(m0 + e) * OUT_F + n] = acc[i][j][e] + bv;
        }
    }
}

extern "C" void kernel_launch(void* const* d_in, const int* in_sizes, int n_in,
                              void* d_out, int out_size, void* d_ws, size_t ws_size,
                              hipStream_t stream) {
    const float*    x   = (const float*)d_in[0];
    const uint32_t* qw  = (const uint32_t*)d_in[1];
    const uint32_t* qz  = (const uint32_t*)d_in[2];
    const _Float16* sc  = (const _Float16*)d_in[3];
    const float*    bs  = (const float*)d_in[4];
    float*          out = (float*)d_out;

    const int M = in_sizes[0] / IN_F;   // 4096
    dim3 grid(OUT_F / TN, M / TM);      // (86, 32)
    gptq_wmma_f16_kernel<<<grid, 256, 0, stream>>>(x, qw, qz, sc, bs, out);
}
